// HeteroGCN_13846974562751
// MI455X (gfx1250) — matrix operand order repaired and checked by reading, not compiled, not run on hardware
//
#include <hip/hip_runtime.h>
#include <hip/hip_bf16.h>
#include <stdint.h>

// ---------------------------------------------------------------------------
// Types for CDNA5 WMMA
// ---------------------------------------------------------------------------
typedef __attribute__((ext_vector_type(4)))  __bf16 v4bf;
typedef __attribute__((ext_vector_type(8)))  __bf16 v8bf;
typedef __attribute__((ext_vector_type(16))) __bf16 v16bf;
typedef __attribute__((ext_vector_type(8)))  float  v8f;

__device__ __forceinline__ __bf16 f2bf(float f) {
  // round-to-nearest-even f32 -> bf16 (only used in bandwidth-bound pre-passes)
  unsigned u = __builtin_bit_cast(unsigned, f);
  u += 0x7FFFu + ((u >> 16) & 1u);
  unsigned short h = (unsigned short)(u >> 16);
  return __builtin_bit_cast(__bf16, h);
}

__device__ __forceinline__ v16bf cat8(v8bf lo, v8bf hi) {
  return __builtin_shufflevector(lo, hi, 0, 1, 2, 3, 4, 5, 6, 7,
                                 8, 9, 10, 11, 12, 13, 14, 15);
}

// ---------------------------------------------------------------------------
// Zero fill
// ---------------------------------------------------------------------------
__global__ void zero_f32(float* __restrict__ p, size_t n) {
  size_t i = (size_t)blockIdx.x * blockDim.x + threadIdx.x;
  if (i < n) p[i] = 0.0f;
}

// ---------------------------------------------------------------------------
// Degree accumulation + inverse-sqrt finalize
// ---------------------------------------------------------------------------
__global__ void deg_count(const int* __restrict__ src, const int* __restrict__ dst,
                          float* __restrict__ dout, float* __restrict__ din, int E) {
  int e = blockIdx.x * blockDim.x + threadIdx.x;
  if (e < E) {
    atomicAdd(&dout[src[e]], 1.0f);
    atomicAdd(&din[dst[e]], 1.0f);
  }
}

__global__ void rsqrt_clamp(float* __restrict__ d, size_t n) {
  size_t i = (size_t)blockIdx.x * blockDim.x + threadIdx.x;
  if (i < n) d[i] = rsqrtf(fmaxf(d[i], 1.0f));
}

// ---------------------------------------------------------------------------
// Pre-pass 1: Y[M x 128] (bf16, row-major) = bf16( X * s[row] )
// ---------------------------------------------------------------------------
__global__ void scale_cvt_bf16(const float* __restrict__ X, const float* __restrict__ s,
                               __bf16* __restrict__ Y, size_t total) {
  size_t i = ((size_t)blockIdx.x * blockDim.x + threadIdx.x) * 4;
  if (i >= total) return;
  const float4 v = *(const float4*)(X + i);
  const float sc = s[i >> 7];
  v4bf o;
  o[0] = f2bf(v.x * sc);
  o[1] = f2bf(v.y * sc);
  o[2] = f2bf(v.z * sc);
  o[3] = f2bf(v.w * sc);
  *(v4bf*)(Y + i) = o;
}

// ---------------------------------------------------------------------------
// Pre-pass 2: WT[N2 x 128] (bf16) = bf16( W^T )   (W: [128 x N2] f32)
// ---------------------------------------------------------------------------
__global__ void transpose_cvt_w(const float* __restrict__ W, __bf16* __restrict__ WT,
                                int N2) {
  int idx = blockIdx.x * blockDim.x + threadIdx.x;
  if (idx >= N2 * 128) return;
  int n = idx >> 7;
  int k = idx & 127;
  WT[idx] = f2bf(W[(size_t)k * N2 + n]);
}

// ---------------------------------------------------------------------------
// H[M x N2] = Xbf @ WT^T  via v_wmma_f32_16x16x32_bf16, f32 accumulate.
// One wave computes a 16 x (16*NT) strip.  N2 is a compile-time constant so
// every X/W/H access is base-register + immediate offset; per k-step all
// NT B-fragments are loaded into distinct registers BEFORE the WMMA chain so
// the 8 b128 loads issue as one clause and waits overlap.
//
// A frag (16-bit A 16x32, ISA 7.12.2): lane l -> M=l&15, half=l>>4;
//   elem j<8 : K = j + 8*half ; elem j>=8 : K = 16 + (j-8) + 8*half
// B frag: lane l -> N=l&15, half=l>>4; same K mapping as A frag
// D frag: VGPR r -> M = r + 8*half, N = l&15
// ---------------------------------------------------------------------------
template <int NT, int N2>
__global__ void gemm_wmma_bf16(const __bf16* __restrict__ Xbf,  // [M x 128] bf16
                               const __bf16* __restrict__ WT,   // [N2 x 128] bf16
                               float* __restrict__ H,           // [M x N2] f32
                               int M) {
  constexpr int NGROUPS = N2 / (16 * NT);
  const int lane = threadIdx.x & 31;
  const int wave = threadIdx.x >> 5;
  const int tile = blockIdx.x * (blockDim.x >> 5) + wave;
  const int mt = tile / NGROUPS;
  const int ng = tile - mt * NGROUPS;
  if (mt * 16 >= M) return;

  const int half = lane >> 4;
  const int l15  = lane & 15;
  const int row  = mt * 16 + l15;
  // A base: this lane's row, lane-half K offset; k-step adds immediates.
  const __bf16* abase = Xbf + ((size_t)row << 7) + 8 * half;
  // B base: column (ng*NT*16 + l15), lane-half K offset; t/kk add immediates
  // (t*16 columns -> t<<11 elements; max byte offset ~12.5KB < 24-bit ioffset)
  const __bf16* wbase = WT + ((size_t)(ng * NT * 16 + l15) << 7) + 8 * half;

  v8f acc[NT];
#pragma unroll
  for (int t = 0; t < NT; ++t) acc[t] = (v8f){};

#pragma unroll
  for (int kk = 0; kk < 128; kk += 32) {
    const v16bf a = cat8(*(const v8bf*)(abase + kk),
                         *(const v8bf*)(abase + kk + 16));
    v16bf bfrag[NT];
#pragma unroll
    for (int t = 0; t < NT; ++t) {
      const __bf16* wp = wbase + (t << 11) + kk;
      bfrag[t] = cat8(*(const v8bf*)wp, *(const v8bf*)(wp + 16));
    }
#pragma unroll
    for (int t = 0; t < NT; ++t) {
      acc[t] = __builtin_amdgcn_wmma_f32_16x16x32_bf16(false, a, false, bfrag[t],
                                                       (short)0, acc[t],
                                                       false, false);
    }
  }

  // D store: base + compile-time offsets (stride N2 is constexpr)
  float* hbase = H + (size_t)(mt * 16 + 8 * half) * N2 + ng * NT * 16 + l15;
#pragma unroll
  for (int t = 0; t < NT; ++t) {
#pragma unroll
    for (int r = 0; r < 8; ++r) {
      hbase[(size_t)r * N2 + t * 16] = acc[t][r];
    }
  }
}

// ---------------------------------------------------------------------------
// SpMM scatter: agg[dst[e]] += h[src[e]]   (D floats per node, f32 atomics)
// ---------------------------------------------------------------------------
template <int D>
__global__ void spmm_scatter(const float* __restrict__ h,
                             const int* __restrict__ src,
                             const int* __restrict__ dst,
                             float* __restrict__ agg, int E) {
  constexpr int TPE = D / 4;
  const long long gid = (long long)blockIdx.x * blockDim.x + threadIdx.x;
  const int e = (int)(gid / TPE);
  const int c = (int)(gid % TPE);
  if (e >= E) return;
  const int si = src[e];
  const int di = dst[e];
  const float4 v = *(const float4*)(h + (size_t)si * D + c * 4);
  float* p = agg + (size_t)di * D + c * 4;
  atomicAdd(p + 0, v.x);
  atomicAdd(p + 1, v.y);
  atomicAdd(p + 2, v.z);
  atomicAdd(p + 3, v.w);
}

// ---------------------------------------------------------------------------
// Combine: normalize by deg_in^-1/2, add bias, optional relu
// ---------------------------------------------------------------------------
__global__ void combine1(const float* __restrict__ agg, const float* __restrict__ din,
                         const float* __restrict__ bias, float* __restrict__ out,
                         int N, int D, int relu) {
  const int i = blockIdx.x * blockDim.x + threadIdx.x;
  if (i >= N * D) return;
  const int r = i / D;
  const int j = i - r * D;
  float v = agg[i] * din[r] + bias[j];
  out[i] = relu ? fmaxf(v, 0.0f) : v;
}

__global__ void combine2(const float* __restrict__ a1, const float* __restrict__ d1,
                         const float* __restrict__ b1,
                         const float* __restrict__ a2, const float* __restrict__ d2,
                         const float* __restrict__ b2,
                         float* __restrict__ out, int N, int D, int relu) {
  const int i = blockIdx.x * blockDim.x + threadIdx.x;
  if (i >= N * D) return;
  const int r = i / D;
  const int j = i - r * D;
  float v = a1[i] * d1[r] + b1[j] + a2[i] * d2[r] + b2[j];
  out[i] = relu ? fmaxf(v, 0.0f) : v;
}

// ---------------------------------------------------------------------------
// Edge scoring: out[e] = dot(oA[src[e]], oB[dst[e]])  (dim 64)
// wave32 per edge: float2 per lane + 5-step shfl_xor reduction.
// ---------------------------------------------------------------------------
__global__ void edge_dot64(const float* __restrict__ oA, const float* __restrict__ oB,
                           const int* __restrict__ src, const int* __restrict__ dst,
                           float* __restrict__ out, int E) {
  const long long t = (long long)blockIdx.x * blockDim.x + threadIdx.x;
  const int e = (int)(t >> 5);
  const int lane = (int)(t & 31);
  if (e >= E) return;
  const float2 a = *(const float2*)(oA + (size_t)src[e] * 64 + lane * 2);
  const float2 b = *(const float2*)(oB + (size_t)dst[e] * 64 + lane * 2);
  float p = a.x * b.x + a.y * b.y;
#pragma unroll
  for (int off = 16; off > 0; off >>= 1) p += __shfl_xor(p, off, 32);
  if (lane == 0) out[e] = p;
}

// ---------------------------------------------------------------------------
// Host-side orchestration
// ---------------------------------------------------------------------------
static inline unsigned grid1(size_t n, int blk) { return (unsigned)((n + blk - 1) / blk); }

extern "C" void kernel_launch(void* const* d_in, const int* in_sizes, int n_in,
                              void* d_out, int out_size, void* d_ws, size_t ws_size,
                              hipStream_t stream) {
  (void)n_in; (void)out_size; (void)ws_size;

  // ---- inputs (setup_inputs dict order) ----
  const float* feat_A = (const float*)d_in[0];
  const float* feat_B = (const float*)d_in[1];
  const int* rates_src   = (const int*)d_in[2];
  const int* rates_dst   = (const int*)d_in[3];
  const int* ratedby_src = (const int*)d_in[4];
  const int* ratedby_dst = (const int*)d_in[5];
  const int* follows_src = (const int*)d_in[6];
  const int* follows_dst = (const int*)d_in[7];
  const int* neg_src     = (const int*)d_in[8];
  const int* neg_dst     = (const int*)d_in[9];
  const float* W1_rates   = (const float*)d_in[10];
  const float* b1_rates   = (const float*)d_in[11];
  const float* W1_ratedby = (const float*)d_in[12];
  const float* b1_ratedby = (const float*)d_in[13];
  const float* W1_follows = (const float*)d_in[14];
  const float* b1_follows = (const float*)d_in[15];
  const float* W2_rates   = (const float*)d_in[16];
  const float* b2_rates   = (const float*)d_in[17];
  const float* W2_ratedby = (const float*)d_in[18];
  const float* b2_ratedby = (const float*)d_in[19];
  const float* W2_follows = (const float*)d_in[20];
  const float* b2_follows = (const float*)d_in[21];

  const int IN = 128, HID = 128, OUT = 64;
  const size_t NA = (size_t)in_sizes[0] / IN;
  const size_t NB = (size_t)in_sizes[1] / IN;
  const int E = in_sizes[2];
  float* pos_out = (float*)d_out;
  float* neg_out = pos_out + E;

  // ---- workspace layout (float units), layer-2 aliases dead layer-1 ----
  float* ws = (float*)d_ws;
  size_t off = 0;
  auto alloc = [&](size_t n) { float* p = ws + off; off += (n + 63) & ~(size_t)63; return p; };

  float* dro_rates = alloc(NA);
  float* dri_rates = alloc(NB);
  float* dro_rb    = alloc(NB);
  float* dri_rb    = alloc(NA);
  float* dro_fo    = alloc(NA);
  float* dri_fo    = alloc(NA);
  const size_t degTotal = off;

  float* h_rates = alloc(NA * HID);
  float* h_rb    = alloc(NB * HID);
  float* h_fo    = alloc(NA * HID);
  float* agg_rates = alloc(NB * HID);
  float* agg_rb    = alloc(NA * HID);
  float* agg_fo    = alloc(NA * HID);
  float* hA = alloc(NA * HID);
  float* hB = alloc(NB * HID);

  const size_t NMAX = NA > NB ? NA : NB;
  __bf16* Xbf = (__bf16*)alloc(NMAX * 64);  // NMAX*128 bf16
  __bf16* WTb = (__bf16*)alloc(128 * 64);   // 128*128 bf16

  float* h2_rates = h_rates;
  float* h2_rb    = h_rb;
  float* h2_fo    = h_fo;
  float* agg2_rates = agg_rates;
  float* agg2_rb    = agg_rb;
  float* agg2_fo    = agg_fo;
  float* oA = hA;
  float* oB = hB;

  const int BLK = 256;

  // one GEMM = scale+cvt X -> bf16, transpose+cvt W -> bf16, WMMA gemm
  auto run_gemm128 = [&](const float* X, const float* s, const float* W, float* H,
                         size_t M) {
    scale_cvt_bf16<<<grid1(M * 32, BLK), BLK, 0, stream>>>(X, s, Xbf, M * 128);
    transpose_cvt_w<<<grid1((size_t)128 * 128, BLK), BLK, 0, stream>>>(W, WTb, 128);
    const size_t tiles = (M / 16) * 2;  // NT=4 -> 64 cols/wave, 2 groups
    gemm_wmma_bf16<4, 128><<<grid1(tiles, 8), BLK, 0, stream>>>(Xbf, WTb, H, (int)M);
  };
  auto run_gemm64 = [&](const float* X, const float* s, const float* W, float* H,
                        size_t M) {
    scale_cvt_bf16<<<grid1(M * 32, BLK), BLK, 0, stream>>>(X, s, Xbf, M * 128);
    transpose_cvt_w<<<grid1((size_t)64 * 128, BLK), BLK, 0, stream>>>(W, WTb, 64);
    const size_t tiles = (M / 16);      // NT=4 -> 64 cols/wave, 1 group
    gemm_wmma_bf16<4, 64><<<grid1(tiles, 8), BLK, 0, stream>>>(Xbf, WTb, H, (int)M);
  };

  // ---- degrees ----
  zero_f32<<<grid1(degTotal, BLK), BLK, 0, stream>>>(ws, degTotal);
  deg_count<<<grid1(E, BLK), BLK, 0, stream>>>(rates_src, rates_dst, dro_rates, dri_rates, E);
  deg_count<<<grid1(E, BLK), BLK, 0, stream>>>(ratedby_src, ratedby_dst, dro_rb, dri_rb, E);
  deg_count<<<grid1(E, BLK), BLK, 0, stream>>>(follows_src, follows_dst, dro_fo, dri_fo, E);
  rsqrt_clamp<<<grid1(degTotal, BLK), BLK, 0, stream>>>(ws, degTotal);

  // ---- layer 1: scaled GEMMs (bf16 WMMA) ----
  run_gemm128(feat_A, dro_rates, W1_rates,   h_rates, NA);
  run_gemm128(feat_B, dro_rb,    W1_ratedby, h_rb,    NB);
  run_gemm128(feat_A, dro_fo,    W1_follows, h_fo,    NA);

  // ---- layer 1: SpMM scatter ----
  zero_f32<<<grid1(NB * HID, BLK), BLK, 0, stream>>>(agg_rates, NB * HID);
  zero_f32<<<grid1(NA * HID, BLK), BLK, 0, stream>>>(agg_rb, NA * HID);
  zero_f32<<<grid1(NA * HID, BLK), BLK, 0, stream>>>(agg_fo, NA * HID);
  spmm_scatter<128><<<grid1((size_t)E * 32, BLK), BLK, 0, stream>>>(h_rates, rates_src, rates_dst, agg_rates, E);
  spmm_scatter<128><<<grid1((size_t)E * 32, BLK), BLK, 0, stream>>>(h_rb, ratedby_src, ratedby_dst, agg_rb, E);
  spmm_scatter<128><<<grid1((size_t)E * 32, BLK), BLK, 0, stream>>>(h_fo, follows_src, follows_dst, agg_fo, E);

  // ---- layer 1: normalize + bias + relu ----
  combine1<<<grid1(NB * HID, BLK), BLK, 0, stream>>>(agg_rates, dri_rates, b1_rates, hB, (int)NB, HID, 1);
  combine2<<<grid1(NA * HID, BLK), BLK, 0, stream>>>(agg_rb, dri_rb, b1_ratedby,
                                                     agg_fo, dri_fo, b1_follows,
                                                     hA, (int)NA, HID, 1);

  // ---- layer 2: scaled GEMMs (bf16 WMMA, N2 = 64) ----
  run_gemm64(hA, dro_rates, W2_rates,   h2_rates, NA);
  run_gemm64(hB, dro_rb,    W2_ratedby, h2_rb,    NB);
  run_gemm64(hA, dro_fo,    W2_follows, h2_fo,    NA);

  // ---- layer 2: SpMM scatter ----
  zero_f32<<<grid1(NB * OUT, BLK), BLK, 0, stream>>>(agg2_rates, NB * OUT);
  zero_f32<<<grid1(NA * OUT, BLK), BLK, 0, stream>>>(agg2_rb, NA * OUT);
  zero_f32<<<grid1(NA * OUT, BLK), BLK, 0, stream>>>(agg2_fo, NA * OUT);
  spmm_scatter<64><<<grid1((size_t)E * 16, BLK), BLK, 0, stream>>>(h2_rates, rates_src, rates_dst, agg2_rates, E);
  spmm_scatter<64><<<grid1((size_t)E * 16, BLK), BLK, 0, stream>>>(h2_rb, ratedby_src, ratedby_dst, agg2_rb, E);
  spmm_scatter<64><<<grid1((size_t)E * 16, BLK), BLK, 0, stream>>>(h2_fo, follows_src, follows_dst, agg2_fo, E);

  // ---- layer 2: normalize + bias (no relu) ----
  combine1<<<grid1(NB * OUT, BLK), BLK, 0, stream>>>(agg2_rates, dri_rates, b2_rates, oB, (int)NB, OUT, 0);
  combine2<<<grid1(NA * OUT, BLK), BLK, 0, stream>>>(agg2_rb, dri_rb, b2_ratedby,
                                                     agg2_fo, dri_fo, b2_follows,
                                                     oA, (int)NA, OUT, 0);

  // ---- scoring ----
  edge_dot64<<<grid1((size_t)E * 32, BLK), BLK, 0, stream>>>(oA, oB, rates_src, rates_dst, pos_out, E);
  edge_dot64<<<grid1((size_t)E * 32, BLK), BLK, 0, stream>>>(oA, oB, neg_src, neg_dst, neg_out, E);
}
